// new_18562848653767
// MI455X (gfx1250) — matrix operand order repaired and checked by reading, not compiled, not run on hardware
//
#include <hip/hip_runtime.h>

// ---------------- constants ----------------
#define kB   8
#define kC   24
#define kA   96
#define kW   96
#define kF   32
#define kNH  128
#define kBC  (kB*kC)            // 192
#define kPIX (kA*kW)            // 9216 pixels per (b,c) matrix
#define kTIL (kPIX/16)          // 576 tiles per matrix
#define kTILES_TOTAL (kBC*kTIL) // 110592
#define kGEMM_BLOCKS (kTILES_TOTAL/8) // 13824 blocks of 8 waves

typedef __attribute__((ext_vector_type(16))) __bf16 v16bf;
typedef __attribute__((ext_vector_type(8)))  float  v8f;

union Frag { v16bf v; unsigned u[8]; };

// ---------------- helpers ----------------
__device__ __forceinline__ unsigned short f2bf(float f) {
    unsigned u = __float_as_uint(f);
    unsigned r = u + 0x7FFFu + ((u >> 16) & 1u);   // RNE
    return (unsigned short)(r >> 16);
}
__device__ __forceinline__ float bf2f(unsigned short s) {
    return __uint_as_float(((unsigned)s) << 16);
}
__device__ __forceinline__ unsigned pack2bf(float a, float b) {
    return (unsigned)f2bf(a) | ((unsigned)f2bf(b) << 16);
}
// hardware sin: v_sin_f32 takes revolutions (x/2pi)
__device__ __forceinline__ float sinhw(float x) {
    return __builtin_amdgcn_sinf(x * 0.15915494309189535f);
}
// order-preserving float->uint for max reductions
__device__ __forceinline__ unsigned ford(float f) {
    unsigned u = __float_as_uint(f);
    return (u & 0x80000000u) ? ~u : (u | 0x80000000u);
}
__device__ __forceinline__ float funord(unsigned e) {
    unsigned u = (e & 0x80000000u) ? (e ^ 0x80000000u) : ~e;
    return __uint_as_float(u);
}
__device__ __forceinline__ v8f wmma_bf16(const Frag& a, const Frag& b, v8f c) {
    return __builtin_amdgcn_wmma_f32_16x16x32_bf16(false, a.v, false, b.v,
                                                   (short)0, c, false, false);
}
__device__ __forceinline__ void loadrow32(const unsigned short* hrow, float* hv) {
    const uint4* r = (const uint4*)hrow;
#pragma unroll
    for (int i = 0; i < 4; ++i) {
        uint4 q = r[i];
        unsigned wds[4] = {q.x, q.y, q.z, q.w};
#pragma unroll
        for (int j = 0; j < 4; ++j) {
            hv[i*8 + 2*j    ] = bf2f((unsigned short)(wds[j] & 0xFFFFu));
            hv[i*8 + 2*j + 1] = bf2f((unsigned short)(wds[j] >> 16));
        }
    }
}

// ---------------- weight conversion (fp32 -> bf16, transposed [N][K]) ----------------
__global__ void k_convert(const float* __restrict__ t0w2, const float* __restrict__ tw1,
                          const float* __restrict__ tw2,
                          unsigned short* __restrict__ o0,   // [32][128]
                          unsigned short* __restrict__ o1,   // [4][128][64]
                          unsigned short* __restrict__ o2) { // [4][32][128]
    int i = blockIdx.x * 256 + threadIdx.x;
    if (i < 32*128) {
        int n = i / 128, k = i % 128;
        o0[n*128 + k] = f2bf(t0w2[k*32 + n]);
    } else if (i < 32*128 + 4*128*64) {
        int j = i - 32*128;
        int s = j / (128*64), r = j % (128*64);
        int n = r / 64, k = r % 64;
        o1[s*8192 + n*64 + k] = f2bf(tw1[s*8192 + k*128 + n]);
    } else if (i < 32*128 + 4*128*64 + 4*32*128) {
        int j = i - 32*128 - 4*128*64;
        int s = j / (32*128), r = j % (32*128);
        int n = r / 128, k = r % 128;
        o2[s*4096 + n*128 + k] = f2bf(tw2[s*4096 + k*32 + n]);
    }
}

// ---------------- per-(b,c) masked mean / inv-std / count ----------------
__global__ void __launch_bounds__(256) k_stats(const float* __restrict__ x,
                                               const int* __restrict__ rl,
                                               const int* __restrict__ cl,
                                               float4* __restrict__ stats) {
    __shared__ float s1[256], s2[256];
    int bc = blockIdx.x, t = threadIdx.x;
    int rlen = rl[bc], clen = cl[bc];
    const float* xm = x + (size_t)bc * kPIX;
    float s = 0.f, q = 0.f;
    for (int p = t; p < kPIX; p += 256) {
        int a = p / kW, w = p % kW;
        if (a < rlen && w < clen) { float v = xm[p]; s += v; q += v*v; }
    }
    s1[t] = s; s2[t] = q; __syncthreads();
    for (int o = 128; o > 0; o >>= 1) {
        if (t < o) { s1[t] += s1[t+o]; s2[t] += s2[t+o]; }
        __syncthreads();
    }
    if (t == 0) {
        float n = (float)(rlen * clen);
        float mean = s1[0] / n;
        float var  = fmaxf((s2[0] - n*mean*mean) / fmaxf(n - 1.f, 1.f), 0.f);
        float sd   = fmaxf(sqrtf(var), 1e-12f);
        stats[bc]  = make_float4(mean, 1.f/sd, n, 0.f);
    }
}

// ---------------- t0 MLP: xn -> relu(xn*w1+b1) [16x128] -> @w2 [128x32] ----------------
__global__ void __launch_bounds__(256) k_t0(const float* __restrict__ x,
                                            const int* __restrict__ rl, const int* __restrict__ cl,
                                            const float4* __restrict__ stats,
                                            const float* __restrict__ w1, const float* __restrict__ b1,
                                            const unsigned short* __restrict__ w2t, // [32][128]
                                            const float* __restrict__ b2,
                                            unsigned short* __restrict__ hbuf) {
    int lane = threadIdx.x & 31, wid = threadIdx.x >> 5;
    int tile = blockIdx.x * 8 + wid;
    int bc = tile / kTIL, t = tile % kTIL;
    int rlen = rl[bc], clen = cl[bc];
    float4 st = stats[bc];
    int m = lane & 15, half = lane >> 4;
    int lp = t*16 + m;
    int a = lp / kW, w = lp % kW;
    float msk = (a < rlen && w < clen) ? 1.f : 0.f;
    float xn = (x[(size_t)bc * kPIX + lp] - st.x) * st.y * msk;

    // Activation A-fragments: act[m][k] = relu(xn*w1[k]+b1[k]), built in-register per lane
    Frag af[4];
#pragma unroll
    for (int ks = 0; ks < 4; ++ks)
#pragma unroll
        for (int v = 0; v < 8; ++v) {
            int k0 = ks*32 + (v < 4 ? 0 : 16) + half*8 + (v & 3)*2;
            float a0 = fmaxf(xn * w1[k0]     + b1[k0],     0.f);
            float a1 = fmaxf(xn * w1[k0 + 1] + b1[k0 + 1], 0.f);
            af[ks].u[v] = pack2bf(a0, a1);
        }

#pragma unroll
    for (int nt = 0; nt < 2; ++nt) {
        int n = nt*16 + m;
        Frag bfr[4];
#pragma unroll
        for (int ks = 0; ks < 4; ++ks)
#pragma unroll
            for (int v = 0; v < 8; ++v) {
                int k0 = ks*32 + (v < 4 ? 0 : 16) + half*8 + (v & 3)*2;
                bfr[ks].u[v] = *(const unsigned*)(w2t + n*128 + k0);
            }
        v8f acc = {0.f,0.f,0.f,0.f,0.f,0.f,0.f,0.f};
#pragma unroll
        for (int ks = 0; ks < 4; ++ks) acc = wmma_bf16(af[ks], bfr[ks], acc);
        float bias = b2[n];
#pragma unroll
        for (int v = 0; v < 8; ++v) {
            int mr = v + 8*half, lpr = t*16 + mr;
            int ar = lpr / kW, wr = lpr % kW;
            float mk = (ar < rlen && wr < clen) ? 1.f : 0.f;
            hbuf[(size_t)(bc * kPIX + lpr) * kF + n] = f2bf((acc[v] + bias) * mk);
        }
    }
}

// ---------------- pools per (b,c): e1(a,j), e2(w,j), e3(j), partial max over (a,w) ----------------
__global__ void __launch_bounds__(256) k_pool(const unsigned short* __restrict__ hbuf,
                                              float* __restrict__ e1, float* __restrict__ e2,
                                              float* __restrict__ e3, unsigned* __restrict__ cmax) {
    __shared__ float e1s[kA][8];
    __shared__ float e2s[kW][8];
    __shared__ float e3s[8];
    __shared__ unsigned cms[kF];
    int bc = blockIdx.x, t = threadIdx.x;
    for (int i = t; i < kA*8; i += 256) { e1s[i/8][i%8] = 0.f; e2s[i/8][i%8] = 0.f; }
    if (t < 8)  e3s[t] = 0.f;
    if (t < kF) cms[t] = 0u;
    __syncthreads();

    float e3p[8] = {0.f,0.f,0.f,0.f,0.f,0.f,0.f,0.f};
    unsigned mx[kF];
#pragma unroll
    for (int f = 0; f < kF; ++f) mx[f] = 0u;

    const unsigned short* hb = hbuf + (size_t)bc * kPIX * kF;
    for (int p = t; p < kPIX; p += 256) {
        if (p + 256 < kPIX) __builtin_prefetch(hb + (size_t)(p + 256) * kF, 0, 1);
        int a = p / kW, w = p % kW;
        float hv[kF];
        loadrow32(hb + (size_t)p * kF, hv);
#pragma unroll
        for (int f = 0; f < kF; ++f) { unsigned e = ford(hv[f]); mx[f] = (e > mx[f]) ? e : mx[f]; }
#pragma unroll
        for (int j = 0; j < 8; ++j) {
            atomicAdd(&e1s[a][j], sinhw(hv[4*j + 1]));
            atomicAdd(&e2s[w][j], sinhw(hv[4*j + 2]));
            e3p[j] += sinhw(hv[4*j + 3]);
        }
    }
#pragma unroll
    for (int j = 0; j < 8; ++j) atomicAdd(&e3s[j], e3p[j]);
#pragma unroll
    for (int f = 0; f < kF; ++f) atomicMax(&cms[f], mx[f]);
    __syncthreads();

    for (int i = t; i < kA*8; i += 256) { e1[bc*kA*8 + i] = e1s[i/8][i%8]; e2[bc*kW*8 + i] = e2s[i/8][i%8]; }
    if (t < 8)  e3[bc*8 + t] = e3s[t];
    if (t < kF) cmax[bc*kF + t] = cms[t];
}

// ---------------- cross-C max -> per-batch h1 broadcast ----------------
__global__ void k_bmax(const unsigned* __restrict__ cmax, float* __restrict__ h1g) {
    int b = blockIdx.x, f = threadIdx.x;
    unsigned e = 0u;
    for (int c = 0; c < kC; ++c) {
        unsigned v = cmax[(b*kC + c)*kF + f];
        e = (v > e) ? v : e;
    }
    h1g[b*kF + f] = funord(e);
}

// ---------------- stack update: hp[16x64] @ w1[64x128] -> relu -> @ w2[128x32] + skip ----------------
__global__ void __launch_bounds__(256) k_stack(const int* __restrict__ rl, const int* __restrict__ cl,
                                               const unsigned short* __restrict__ w1t, // [128][64]
                                               const float* __restrict__ b1,
                                               const unsigned short* __restrict__ w2t, // [32][128]
                                               const float* __restrict__ b2,
                                               const float* __restrict__ e1, const float* __restrict__ e2,
                                               const float* __restrict__ e3, const float* __restrict__ h1g,
                                               unsigned short* __restrict__ hbuf) {
    __shared__ unsigned short pre[8][16][kNH]; // per-wave relu(GEMM1) tiles, bf16
    int lane = threadIdx.x & 31, wid = threadIdx.x >> 5;
    int tile = blockIdx.x * 8 + wid;
    int bc = tile / kTIL, t = tile % kTIL;
    int b  = bc / kC;
    int rlen = rl[bc], clen = cl[bc];
    int m = lane & 15, half = lane >> 4;
    int lp = t*16 + m;
    int a = lp / kW, w = lp % kW;

    unsigned short* hb = hbuf + (size_t)bc * kPIX * kF;
    const float* e1r = e1 + bc*kA*8 + a*8;
    const float* e2r = e2 + bc*kW*8 + w*8;
    const float* e3r = e3 + bc*8;
    const float* h1r = h1g + b*kF;

    auto hpElem = [&](int k) -> float {
        if (k < 8)  return sinhw(bf2f(hb[(size_t)lp * kF + 4*k]));  // e0 (h already masked)
        if (k < 16) return e1r[k - 8];
        if (k < 24) return e2r[k - 16];
        if (k < 32) return e3r[k - 24];
        return h1r[k - 32];
    };

    // A-fragments for hp (16x64), K=0..63 over 2 steps
    Frag a1[2];
#pragma unroll
    for (int ks = 0; ks < 2; ++ks)
#pragma unroll
        for (int v = 0; v < 8; ++v) {
            int k0 = ks*32 + (v < 4 ? 0 : 16) + half*8 + (v & 3)*2;
            a1[ks].u[v] = pack2bf(hpElem(k0), hpElem(k0 + 1));
        }

    // GEMM1: (16x64) @ (64x128), relu+bias -> pre (LDS)
#pragma unroll
    for (int nt = 0; nt < 8; ++nt) {
        int n = nt*16 + m;
        Frag bfr[2];
#pragma unroll
        for (int ks = 0; ks < 2; ++ks)
#pragma unroll
            for (int v = 0; v < 8; ++v) {
                int k0 = ks*32 + (v < 4 ? 0 : 16) + half*8 + (v & 3)*2;
                bfr[ks].u[v] = *(const unsigned*)(w1t + n*64 + k0);
            }
        v8f acc = {0.f,0.f,0.f,0.f,0.f,0.f,0.f,0.f};
        acc = wmma_bf16(a1[0], bfr[0], acc);
        acc = wmma_bf16(a1[1], bfr[1], acc);
        float bias = b1[n];
#pragma unroll
        for (int v = 0; v < 8; ++v)
            pre[wid][v + 8*half][n] = f2bf(fmaxf(acc[v] + bias, 0.f));
    }
    __syncthreads(); // D-layout -> A-layout re-distribution via LDS

    // GEMM2: (16x128) @ (128x32) + bias + skip, masked, in-place store
    Frag a2[4];
#pragma unroll
    for (int ks = 0; ks < 4; ++ks)
#pragma unroll
        for (int v = 0; v < 8; ++v) {
            int k0 = ks*32 + (v < 4 ? 0 : 16) + half*8 + (v & 3)*2;
            a2[ks].u[v] = *(const unsigned*)&pre[wid][m][k0];
        }
#pragma unroll
    for (int nt = 0; nt < 2; ++nt) {
        int n = nt*16 + m;
        Frag bfr[4];
#pragma unroll
        for (int ks = 0; ks < 4; ++ks)
#pragma unroll
            for (int v = 0; v < 8; ++v) {
                int k0 = ks*32 + (v < 4 ? 0 : 16) + half*8 + (v & 3)*2;
                bfr[ks].u[v] = *(const unsigned*)(w2t + n*128 + k0);
            }
        v8f acc = {0.f,0.f,0.f,0.f,0.f,0.f,0.f,0.f};
#pragma unroll
        for (int ks = 0; ks < 4; ++ks) acc = wmma_bf16(a2[ks], bfr[ks], acc);
        float bias = b2[n];
#pragma unroll
        for (int v = 0; v < 8; ++v) {
            int mr = v + 8*half, lpr = t*16 + mr;
            int ar = lpr / kW, wr = lpr % kW;
            float mk = (ar < rlen && wr < clen) ? 1.f : 0.f;
            size_t idx = (size_t)lpr * kF + n;
            float skip = bf2f(hb[idx]);
            hb[idx] = f2bf((skip + acc[v] + bias) * mk);
        }
    }
}

// ---------------- per-(b,c) RMS over valid region ----------------
__global__ void __launch_bounds__(256) k_rms(const unsigned short* __restrict__ hbuf,
                                             const float4* __restrict__ stats,
                                             float* __restrict__ hbc) {
    __shared__ float sums[kF];
    int bc = blockIdx.x, t = threadIdx.x;
    if (t < kF) sums[t] = 0.f;
    __syncthreads();
    const unsigned short* hb = hbuf + (size_t)bc * kPIX * kF;
    float part[kF];
#pragma unroll
    for (int f = 0; f < kF; ++f) part[f] = 0.f;
    for (int p = t; p < kPIX; p += 256) {
        if (p + 256 < kPIX) __builtin_prefetch(hb + (size_t)(p + 256) * kF, 0, 1);
        float hv[kF];
        loadrow32(hb + (size_t)p * kF, hv);
#pragma unroll
        for (int f = 0; f < kF; ++f) part[f] += hv[f] * hv[f];
    }
#pragma unroll
    for (int f = 0; f < kF; ++f) atomicAdd(&sums[f], part[f]);
    __syncthreads();
    if (t < kF) {
        float num = fmaxf(sums[t], 1e-20f);
        float den = fmaxf(stats[bc].z, 1e-12f);
        hbc[bc*kF + t] = sqrtf(num / den);
    }
}

// ---------------- output head: mean over C, MLP 32->128->2, 8*tanh ----------------
__global__ void __launch_bounds__(128) k_final(const float* __restrict__ hbc,
                                               const float* __restrict__ w1, const float* __restrict__ b1,
                                               const float* __restrict__ w2, const float* __restrict__ b2,
                                               float* __restrict__ out) {
    __shared__ float hbar[kF];
    __shared__ float hid[kNH];
    int b = blockIdx.x, t = threadIdx.x;
    if (t < kF) {
        float s = 0.f;
        for (int c = 0; c < kC; ++c) s += hbc[(b*kC + c)*kF + t];
        hbar[t] = s / (float)kC;
    }
    __syncthreads();
    {
        float s = b1[t];
        for (int k = 0; k < kF; ++k) s += hbar[k] * w1[k*kNH + t];
        hid[t] = fmaxf(s, 0.f);
    }
    __syncthreads();
    if (t < 2) {
        float s = b2[t];
        for (int k = 0; k < kNH; ++k) s += hid[k] * w2[k*2 + t];
        out[b*2 + t] = 8.f * tanhf(s);
    }
}

// ---------------- host launch ----------------
extern "C" void kernel_launch(void* const* d_in, const int* in_sizes, int n_in,
                              void* d_out, int out_size, void* d_ws, size_t ws_size,
                              hipStream_t stream) {
    (void)in_sizes; (void)n_in; (void)out_size; (void)ws_size;
    const float* x     = (const float*)d_in[0];
    const int*   rl    = (const int*)  d_in[1];
    const int*   cl    = (const int*)  d_in[2];
    const float* t0_w1 = (const float*)d_in[3];
    const float* t0_b1 = (const float*)d_in[4];
    const float* t0_w2 = (const float*)d_in[5];
    const float* t0_b2 = (const float*)d_in[6];
    const float* t_w1  = (const float*)d_in[7];
    const float* t_b1  = (const float*)d_in[8];
    const float* t_w2  = (const float*)d_in[9];
    const float* t_b2  = (const float*)d_in[10];
    const float* to_w1 = (const float*)d_in[11];
    const float* to_b1 = (const float*)d_in[12];
    const float* to_w2 = (const float*)d_in[13];
    const float* to_b2 = (const float*)d_in[14];

    char* ws = (char*)d_ws;
    size_t off = 0;
    auto alloc = [&](size_t bytes) -> char* {
        char* p = ws + off;
        off = (off + bytes + 255) & ~(size_t)255;
        return p;
    };
    float4*         stats = (float4*)        alloc((size_t)kBC * 16);
    unsigned short* w2t0  = (unsigned short*)alloc((size_t)32*128*2);
    unsigned short* w1ts  = (unsigned short*)alloc((size_t)4*128*64*2);
    unsigned short* w2ts  = (unsigned short*)alloc((size_t)4*32*128*2);
    unsigned short* hbuf  = (unsigned short*)alloc((size_t)kBC * kPIX * kF * 2);
    float*          e1    = (float*)         alloc((size_t)kBC * kA * 8 * 4);
    float*          e2    = (float*)         alloc((size_t)kBC * kW * 8 * 4);
    float*          e3    = (float*)         alloc((size_t)kBC * 8 * 4);
    unsigned*       cmax  = (unsigned*)      alloc((size_t)kBC * kF * 4);
    float*          h1g   = (float*)         alloc((size_t)kB * kF * 4);
    float*          hbc   = (float*)         alloc((size_t)kBC * kF * 4);

    k_convert<<<(32*128 + 4*128*64 + 4*32*128 + 255)/256, 256, 0, stream>>>(
        t0_w2, t_w1, t_w2, w2t0, w1ts, w2ts);
    k_stats<<<kBC, 256, 0, stream>>>(x, rl, cl, stats);
    k_t0<<<kGEMM_BLOCKS, 256, 0, stream>>>(x, rl, cl, stats, t0_w1, t0_b1, w2t0, t0_b2, hbuf);
    for (int i = 0; i < 4; ++i) {
        k_pool<<<kBC, 256, 0, stream>>>(hbuf, e1, e2, e3, cmax);
        k_bmax<<<kB, kF, 0, stream>>>(cmax, h1g);
        k_stack<<<kGEMM_BLOCKS, 256, 0, stream>>>(rl, cl,
            w1ts + (size_t)i*128*64, t_b1 + (size_t)i*128,
            w2ts + (size_t)i*32*128, t_b2 + (size_t)i*32,
            e1, e2, e3, h1g, hbuf);
    }
    k_rms<<<kBC, 256, 0, stream>>>(hbuf, stats, hbc);
    k_final<<<kB, 128, 0, stream>>>(hbc, to_w1, to_b1, to_w2, to_b2, (float*)d_out);
}
